// GRUQR_84688165143039
// MI455X (gfx1250) — matrix operand order repaired and checked by reading, not compiled, not run on hardware
//
#include <hip/hip_runtime.h>
#include <hip/hip_bf16.h>

typedef _Float16 v16h __attribute__((ext_vector_type(16)));
typedef _Float16 v8h  __attribute__((ext_vector_type(8)));
typedef float    v8f  __attribute__((ext_vector_type(8)));

#define NSTEP 30
#define HB    4096          // batch
#define HID   256
#define XC    320           // padded gin width: [s:0..2 |pad| te:32..63 | hf:64..319]

// ---------------- WMMA helpers (CDNA5 16x16x32 f16 -> f32) ----------------

static __device__ __forceinline__ v16h cat8(v8h lo, v8h hi) {
    return __builtin_shufflevector(lo, hi, 0,1,2,3,4,5,6,7,8,9,10,11,12,13,14,15);
}

static __device__ __forceinline__ v8f wmma(v16h a, v16h b, v8f c) {
    return __builtin_amdgcn_wmma_f32_16x16x32_f16(false, a, false, b, (short)0, c, false, false);
}

// A-matrix fragment from LDS (row-major f16, 16 rows). 16-bit A 16x32 layout:
// lane<16: M=lane, K={0..7,16..23}; lane>=16: M=lane-16, K={8..15,24..31}
static __device__ __forceinline__ v16h ldA(const _Float16* smh, int base, int rstride,
                                           int k0, int lane) {
    int m  = lane & 15;
    int hs = lane >> 4;
    const v8h* p = (const v8h*)(smh + base + m * rstride + k0 + hs * 8);
    return cat8(p[0], p[2]);   // +16 halves for second K chunk
}

// B-matrix fragment from global row-major weight W[out][k]. 16-bit B 32x16 layout:
// lane<16: N=lane, K={0..7,16..23}; lane>=16: N=lane-16, K={8..15,24..31}
static __device__ __forceinline__ v16h ldB(const _Float16* __restrict__ W, int kstride,
                                           int n0, int k0, int lane) {
    int col = n0 + (lane & 15);
    int kk  = k0 + (lane >> 4) * 8;
    const v8h* p = (const v8h*)(W + (size_t)col * kstride + kk);
    return cat8(p[0], p[2]);
}

static __device__ __forceinline__ float sigf(float x) { return 1.f / (1.f + __expf(-x)); }
static __device__ __forceinline__ float softplusf(float x) {
    return (x > 20.f) ? x : log1pf(__expf(x));
}

// One 16-wide hidden tile of a GRU cell: 6 gate accumulators kept in registers,
// fused elementwise GRU update in C layout, writes new h (f16) to LDS.
template<int KSLABS>
static __device__ __forceinline__ void gru_tile(
    _Float16* smh, int aBase, int aStride,          // gi input (A matrix in LDS)
    int hCur, int hNxt,                             // h state buffers (stride HID)
    const _Float16* __restrict__ Wi, int WiKS,
    const _Float16* __restrict__ Wh,
    const float* __restrict__ bi, const float* __restrict__ bh,
    int j0, int lane)
{
    int ln = lane & 15, hs = lane >> 4;
    int oR = j0 + ln, oZ = 256 + j0 + ln, oN = 512 + j0 + ln;
    v8f ir = {}, iz = {}, inn = {}, hr = {}, hz = {}, hn = {};
    ir = ir + bi[oR]; iz = iz + bi[oZ]; inn = inn + bi[oN];
    hr = hr + bh[oR]; hz = hz + bh[oZ]; hn  = hn + bh[oN];
    #pragma unroll
    for (int s = 0; s < KSLABS; ++s) {
        v16h a = ldA(smh, aBase, aStride, s * 32, lane);
        ir  = wmma(a, ldB(Wi, WiKS,       j0, s * 32, lane), ir);
        iz  = wmma(a, ldB(Wi, WiKS, 256 + j0, s * 32, lane), iz);
        inn = wmma(a, ldB(Wi, WiKS, 512 + j0, s * 32, lane), inn);
    }
    #pragma unroll
    for (int s = 0; s < 8; ++s) {
        v16h a = ldA(smh, hCur, HID, s * 32, lane);
        hr = wmma(a, ldB(Wh, HID,       j0, s * 32, lane), hr);
        hz = wmma(a, ldB(Wh, HID, 256 + j0, s * 32, lane), hz);
        hn = wmma(a, ldB(Wh, HID, 512 + j0, s * 32, lane), hn);
    }
    #pragma unroll
    for (int g = 0; g < 8; ++g) {
        int m = g + 8 * hs;
        float hold = (float)smh[hCur + m * HID + j0 + ln];
        float r = sigf(ir[g] + hr[g]);
        float z = sigf(iz[g] + hz[g]);
        float n = tanhf(inn[g] + r * hn[g]);
        smh[hNxt + m * HID + j0 + ln] = (_Float16)((1.f - z) * n + z * hold);
    }
}

// ---------------- setup kernels (f32 -> prepared f16 layouts) ----------------

__global__ void k_convcopy(_Float16* dst, const float* src, int n) {
    int i = blockIdx.x * 256 + threadIdx.x;
    if (i < n) dst[i] = (_Float16)src[i];
}

__global__ void k_wih0(_Float16* dst, const float* W) {  // (768,291) -> (768,320) remapped
    int i = blockIdx.x * 256 + threadIdx.x;
    if (i >= 768 * XC) return;
    int o = i / XC, c = i % XC;
    float v = 0.f;
    if (c < 3)                 v = W[o * 291 + c];            // s
    else if (c >= 32 && c < 64) v = W[o * 291 + 3 + (c - 32)]; // te
    else if (c >= 64)           v = W[o * 291 + 35 + (c - 64)];// h_fused
    dst[i] = (_Float16)v;
}

__global__ void k_wout(_Float16* dst, const float* W) {  // (3,256) -> (16,256) zero padded
    int i = blockIdx.x * 256 + threadIdx.x;
    if (i >= 16 * HID) return;
    int o = i / HID, c = i % HID;
    dst[i] = (_Float16)(o < 3 ? W[o * HID + c] : 0.f);
}

__global__ void k_te(_Float16* te) {  // 30 x 32 time encodings
    int j = threadIdx.x;
    if (j >= 32) return;
    for (int k = 0; k < NSTEP; ++k) {
        float v;
        if (j == 31) v = (float)k / (float)NSTEP;
        else {
            int i = j >> 1;
            float div = expf((float)(2 * i) * (-logf(10000.f) / 31.f));
            float ang = (float)k * div;
            v = (j & 1) ? cosf(ang) : sinf(ang);
        }
        te[k * 32 + j] = (_Float16)v;
    }
}

// ---------------- persistent fused GRU kernel ----------------

__global__ __launch_bounds__(128)
void k_gru(const float* __restrict__ hf,   const float* __restrict__ inits,
           const float* __restrict__ b1,   const float* __restrict__ b2,
           const float* __restrict__ bih0, const float* __restrict__ bhh0,
           const float* __restrict__ bih1, const float* __restrict__ bhh1,
           const float* __restrict__ bout,
           const _Float16* __restrict__ W1p,   const _Float16* __restrict__ W2p,
           const _Float16* __restrict__ Wih0p, const _Float16* __restrict__ Whh0p,
           const _Float16* __restrict__ Wih1p, const _Float16* __restrict__ Whh1p,
           const _Float16* __restrict__ Woutp, const _Float16* __restrict__ te16,
           float* __restrict__ out)
{
    __shared__ __align__(16) _Float16 smh[16 * XC + 4 * 16 * HID]; // x + h0A/h0B/h1A/h1B
    __shared__ float smq[96];                                      // qbuf[48] + ph[48]
    const int X = 0, H0A = 16 * XC, H0B = H0A + 4096, H1A = H0B + 4096, H1B = H1A + 4096;

    int tid = threadIdx.x, lane = tid & 31, wv = tid >> 5;
    int ln = lane & 15, hs = lane >> 4;
    int r0 = blockIdx.x * 16;

    // ---- prologue: build x = [s | pad | te(k=0) | h_fused] as f16, init ph ----
    for (int i = tid; i < 16 * 32; i += 128)
        smh[X + (i >> 5) * XC + (i & 31)] = (_Float16)0.f;
    for (int i = tid; i < 16 * 32; i += 128)
        smh[X + (i >> 5) * XC + 32 + (i & 31)] = te16[i & 31];
    for (int i = tid; i < 16 * HID; i += 128) {
        int m = i >> 8, c = i & 255;
        smh[X + m * XC + 64 + c] = (_Float16)hf[(size_t)(r0 + m) * HID + c];
    }
    if (tid < 16) {
        int m = tid;
        float s0 = inits[(r0 + m) * 3 + 0];
        float s1 = inits[(r0 + m) * 3 + 1];
        float s2 = inits[(r0 + m) * 3 + 2];
        smh[X + m * XC + 0] = (_Float16)s0;
        smh[X + m * XC + 1] = (_Float16)s1;
        smh[X + m * XC + 2] = (_Float16)s2;
        smq[48 + m * 3 + 0] = s0; smq[48 + m * 3 + 1] = s0; smq[48 + m * 3 + 2] = s0;
    }
    __syncthreads();

    // ---- initial-state MLP: t1 = relu(hf@W1.T+b1) -> h = tanh(t1@W2.T+b2) ----
    for (int t = 0; t < 4; ++t) {            // GEMM1: 16 n-tiles / 4 waves
        int n0 = (wv * 4 + t) * 16;
        v8f acc = {}; acc = acc + b1[n0 + ln];
        #pragma unroll
        for (int s = 0; s < 8; ++s) {
            v16h a = ldA(smh, X, XC, 64 + s * 32, lane);
            acc = wmma(a, ldB(W1p, HID, n0, s * 32, lane), acc);
        }
        #pragma unroll
        for (int g = 0; g < 8; ++g) {
            float v = acc[g] > 0.f ? acc[g] : 0.f;
            smh[H1B + (g + 8 * hs) * HID + n0 + ln] = (_Float16)v;   // t1 scratch
        }
    }
    __syncthreads();
    for (int t = 0; t < 8; ++t) {            // GEMM2: 32 n-tiles / 4 waves
        int n0 = (wv * 8 + t) * 16;
        v8f acc = {}; acc = acc + b2[n0 + ln];
        #pragma unroll
        for (int s = 0; s < 8; ++s) {
            v16h a = ldA(smh, H1B, HID, s * 32, lane);
            acc = wmma(a, ldB(W2p, HID, n0, s * 32, lane), acc);
        }
        int dbase = (n0 < 256) ? H0A : H1A;
        int c0    = (n0 < 256) ? n0  : n0 - 256;
        #pragma unroll
        for (int g = 0; g < 8; ++g)
            smh[dbase + (g + 8 * hs) * HID + c0 + ln] = (_Float16)tanhf(acc[g]);
    }
    __syncthreads();

    int h0c = H0A, h0n_ = H0B, h1c = H1A, h1n_ = H1B;

    // ---- 30 recurrent steps, fully resident in LDS/registers ----
    for (int k = 0; k < NSTEP; ++k) {
        for (int t = 0; t < 4; ++t) {        // GRU layer 0 (gin K=320)
            int j0 = wv * 64 + t * 16;
            gru_tile<10>(smh, X, XC, h0c, h0n_, Wih0p, XC, Whh0p, bih0, bhh0, j0, lane);
        }
        __syncthreads();
        for (int t = 0; t < 4; ++t) {        // GRU layer 1 (input = new h0)
            int j0 = wv * 64 + t * 16;
            gru_tile<8>(smh, h0n_, HID, h1c, h1n_, Wih1p, HID, Whh1p, bih1, bhh1, j0, lane);
        }
        __syncthreads();
        if (wv == 0) {                       // quantile head + state update
            v8f acc = {};
            float bo = (ln < 3) ? bout[ln] : 0.f;
            acc = acc + bo;
            #pragma unroll
            for (int s = 0; s < 8; ++s) {
                v16h a = ldA(smh, h1n_, HID, s * 32, lane);
                acc = wmma(a, ldB(Woutp, HID, 0, s * 32, lane), acc);
            }
            if (ln < 3) {
                #pragma unroll
                for (int g = 0; g < 8; ++g) smq[(g + 8 * hs) * 3 + ln] = acc[g];
            }
            if (lane < 16) {                 // per-row scalar chain (in-wave via LDS)
                int m = lane, row = r0 + m;
                float q0 = smq[m * 3 + 0], q1 = smq[m * 3 + 1], q2 = smq[m * 3 + 2];
                float qlo  = q0;
                float qmed = qlo  + softplusf(q1 - qlo);
                float qhi  = qmed + softplusf(q2 - qmed);
                out[(size_t)row * NSTEP + k]                             = qlo;
                out[(size_t)HB * NSTEP + (size_t)row * NSTEP + k]        = qmed;
                out[(size_t)2 * HB * NSTEP + (size_t)row * NSTEP + k]    = qhi;
                float p0 = smq[48 + m * 3 + 1], p1 = smq[48 + m * 3 + 2], p2 = qmed;
                smq[48 + m * 3 + 0] = p0; smq[48 + m * 3 + 1] = p1; smq[48 + m * 3 + 2] = p2;
                float vv = (p2 - p0) * 5.0f;                 // /(2*DT), DT=0.1
                float aa = (p2 - 2.f * p1 + p0) * 100.0f;    // /DT^2
                smh[X + m * XC + 0] = (_Float16)p2;
                smh[X + m * XC + 1] = (_Float16)vv;
                smh[X + m * XC + 2] = (_Float16)aa;
            }
        } else if (k + 1 < NSTEP) {          // other waves refresh te section
            for (int i = tid - 32; i < 16 * 32; i += 96)
                smh[X + (i >> 5) * XC + 32 + (i & 31)] = te16[(k + 1) * 32 + (i & 31)];
        }
        __syncthreads();
        int tmp = h0c; h0c = h0n_; h0n_ = tmp;
        tmp = h1c; h1c = h1n_; h1n_ = tmp;
    }
}

// ---------------- host ----------------

extern "C" void kernel_launch(void* const* d_in, const int* in_sizes, int n_in,
                              void* d_out, int out_size, void* d_ws, size_t ws_size,
                              hipStream_t stream) {
    const float* hf    = (const float*)d_in[0];
    const float* inits = (const float*)d_in[1];
    const float* W1    = (const float*)d_in[2];
    const float* b1    = (const float*)d_in[3];
    const float* W2    = (const float*)d_in[4];
    const float* b2    = (const float*)d_in[5];
    const float* Wih0  = (const float*)d_in[6];
    const float* Whh0  = (const float*)d_in[7];
    const float* bih0  = (const float*)d_in[8];
    const float* bhh0  = (const float*)d_in[9];
    const float* Wih1  = (const float*)d_in[10];
    const float* Whh1  = (const float*)d_in[11];
    const float* bih1  = (const float*)d_in[12];
    const float* bhh1  = (const float*)d_in[13];
    const float* Wout  = (const float*)d_in[14];
    const float* bout  = (const float*)d_in[15];
    float* out = (float*)d_out;

    char* ws = (char*)d_ws;
    size_t off = 0;
    _Float16* W1p   = (_Float16*)(ws + off); off += 256 * 256 * 2;
    _Float16* W2p   = (_Float16*)(ws + off); off += 512 * 256 * 2;
    _Float16* Wih0p = (_Float16*)(ws + off); off += 768 * 320 * 2;
    _Float16* Whh0p = (_Float16*)(ws + off); off += 768 * 256 * 2;
    _Float16* Wih1p = (_Float16*)(ws + off); off += 768 * 256 * 2;
    _Float16* Whh1p = (_Float16*)(ws + off); off += 768 * 256 * 2;
    _Float16* Woutp = (_Float16*)(ws + off); off += 16 * 256 * 2;
    _Float16* te16  = (_Float16*)(ws + off); off += 30 * 32 * 2;

    k_convcopy<<<(256 * 256 + 255) / 256, 256, 0, stream>>>(W1p,   W1,   256 * 256);
    k_convcopy<<<(512 * 256 + 255) / 256, 256, 0, stream>>>(W2p,   W2,   512 * 256);
    k_convcopy<<<(768 * 256 + 255) / 256, 256, 0, stream>>>(Whh0p, Whh0, 768 * 256);
    k_convcopy<<<(768 * 256 + 255) / 256, 256, 0, stream>>>(Wih1p, Wih1, 768 * 256);
    k_convcopy<<<(768 * 256 + 255) / 256, 256, 0, stream>>>(Whh1p, Whh1, 768 * 256);
    k_wih0<<<(768 * 320 + 255) / 256, 256, 0, stream>>>(Wih0p, Wih0);
    k_wout<<<(16 * 256 + 255) / 256, 256, 0, stream>>>(Woutp, Wout);
    k_te<<<1, 32, 0, stream>>>(te16);

    k_gru<<<HB / 16, 128, 0, stream>>>(hf, inits, b1, b2, bih0, bhh0, bih1, bhh1, bout,
                                       W1p, W2p, Wih0p, Whh0p, Wih1p, Whh1p, Woutp, te16,
                                       out);
}
